// LSTM_70007966925422
// MI455X (gfx1250) — matrix operand order repaired and checked by reading, not compiled, not run on hardware
//
#include <hip/hip_runtime.h>

// ---------------------------------------------------------------------------
// LSTM  (T=512, B=128, I=256, H=256)  for MI455X / gfx1250 (wave32, WMMA).
//
//   1) cvt kernels: fp32 -> bf16 copies of x, W_ih, W_hh (WMMA operands).
//   2) gates GEMM:  gates[T*B,4H] = x_bf16 @ W_ih_bf16^T + b_ih + b_hh
//      (v_wmma_f32_16x16x32_bf16, fp32 accum; memory-bound ~400MB @23.3TB/s).
//   3) persistent recurrence kernel, 16 workgroups (one per 16 hidden cols);
//      W_hh slice staged into LDS via TDM (tensor_load_to_lds, 4 2D tiles);
//      h broadcast via L2; one global atomic barrier per timestep; gates for
//      t+1 prefetched (global_prefetch_b8) at the top of step t.
// ---------------------------------------------------------------------------

#define TT 512
#define BB 128
#define II 256
#define HH 256
#define GG 1024  // 4*H

typedef __attribute__((ext_vector_type(16))) __bf16 v16bf;
typedef __attribute__((ext_vector_type(8)))  float  v8f;

typedef __attribute__((ext_vector_type(4))) unsigned int tdm_v4u;
typedef __attribute__((ext_vector_type(8))) int          tdm_v8i;
typedef __attribute__((ext_vector_type(4))) int          tdm_v4i;

union Frag {            // 16 bf16 = 8 dwords, WMMA A/B operand
    v16bf     v;
    unsigned  u[8];
};

__device__ __forceinline__ float sigmoid_f(float x) {
    return 1.0f / (1.0f + __expf(-x));
}
__device__ __forceinline__ float tanh_f(float x) {
    return 2.0f / (1.0f + __expf(-2.0f * x)) - 1.0f;
}

// ---------------------- TDM: 2D bf16 tile -> LDS ---------------------------
// Builds D# group0/group1 per CDNA5 ISA 8.3/8.4 and issues TENSOR_LOAD_TO_LDS.
// 2D tile: tile_d1 rows of tile_d0 elements (2 bytes each), row stride
// tensor_d0 elements, packed contiguously into LDS at lds_byte.
// Caller must be a single wave (TDM ignores EXEC; one issue per wave) and
// must s_wait_tensorcnt(0) before readers touch the LDS.
// This toolchain exposes the 6-arg builtin:
//   (uint32x4 g0, int32x8 g1, int32x4 g2, int32x4 g3, int32x8 pad, i32 cpol)
__device__ __forceinline__ void tdm_load_2d_to_lds(unsigned lds_byte,
                                                   const void* gptr,
                                                   unsigned tensor_d0,
                                                   unsigned tensor_d1,
                                                   unsigned tile_d0,
                                                   unsigned tile_d1) {
#if __has_builtin(__builtin_amdgcn_tensor_load_to_lds)
    unsigned long long ga = (unsigned long long)(uintptr_t)gptr;
    tdm_v4u g0;
    g0.x = 0x1u;                                   // count=1, load, no gather
    g0.y = lds_byte;                               // D#.lds_addr [63:32]
    g0.z = (unsigned)(ga & 0xffffffffu);           // global_addr [95:64]
    g0.w = (unsigned)((ga >> 32) & 0x01ffffffu)    // global_addr[56:32]
           | (2u << 30);                           // type=2 ("image") [127:126]
    tdm_v8i g1;
    g1[0] = (int)(1u << 16);                       // wg_mask=0, data_size=1 (2B)
    g1[1] = (int)((tensor_d0 & 0xffffu) << 16);    // tensor_dim0[15:0] @ [63:48]
    g1[2] = (int)(((tensor_d0 >> 16) & 0xffffu)    // tensor_dim0[31:16]
                  | ((tensor_d1 & 0xffffu) << 16));// tensor_dim1[15:0] @ [95:80]
    g1[3] = (int)(((tensor_d1 >> 16) & 0xffffu)    // tensor_dim1[31:16]
                  | ((tile_d0 & 0xffffu) << 16));  // tile_dim0 @ [127:112]
    g1[4] = (int)(tile_d1 & 0xffffu);              // tile_dim1; tile_dim2=0
    g1[5] = (int)tensor_d0;                        // tensor_dim0_stride[31:0]
    g1[6] = 0;                                     // stride hi / dim1_stride lo
    g1[7] = 0;
    tdm_v4i z4 = {0, 0, 0, 0};
    tdm_v8i z8 = {0, 0, 0, 0, 0, 0, 0, 0};
    __builtin_amdgcn_tensor_load_to_lds(g0, g1, z4, z4, z8, 0);
#else
    (void)lds_byte; (void)gptr; (void)tensor_d0; (void)tensor_d1;
    (void)tile_d0; (void)tile_d1;
#endif
}

// --------------------------- fp32 -> bf16 copy -----------------------------
__global__ void lstm_cvt_bf16(const float* __restrict__ src,
                              __bf16* __restrict__ dst, int n) {
    for (int i = blockIdx.x * blockDim.x + threadIdx.x; i < n;
         i += gridDim.x * blockDim.x)
        dst[i] = (__bf16)src[i];
}

// ------------------------------- init --------------------------------------
__global__ void lstm_init(const float* __restrict__ h0,
                          const float* __restrict__ c0,
                          __bf16* __restrict__ hbuf,   // [2][B*H], fill buf 0
                          float* __restrict__ cbuf,    // [B*H]
                          unsigned* __restrict__ bar)  // [T] arrive counters
{
    int i = blockIdx.x * blockDim.x + threadIdx.x;
    if (i < BB * HH) {
        hbuf[i] = (__bf16)h0[i];
        cbuf[i] = c0[i];
    }
    if (i < TT) bar[i] = 0u;
}

// ---------------- Phase 1: gates = x @ Wih^T + b_ih + b_hh -----------------
// grid = (M/128, N/64), block = 256 (8 waves). Wave w -> M-tile m0 = w*16.
__global__ __launch_bounds__(256) void lstm_gates_gemm(
    const __bf16* __restrict__ xb,    // [T*B, I]  row-major
    const __bf16* __restrict__ wih,   // [4H, I]   row-major (== B^T)
    const float*  __restrict__ b_ih,  // [4H]
    const float*  __restrict__ b_hh,  // [4H]
    float*        __restrict__ gates) // [T*B, 4H]
{
    __shared__ __bf16 ldsW[64 * II];  // 32 KB: W_ih rows n0..n0+63
    const int tid  = threadIdx.x;
    const int wave = tid >> 5;
    const int lane = tid & 31;
    const int n0   = blockIdx.y * 64;

#if __has_builtin(__builtin_amdgcn_tensor_load_to_lds)
    if (tid < 32) {   // wave 0 only: TDM issues once per wave, ignores EXEC
        tdm_load_2d_to_lds((unsigned)(uintptr_t)&ldsW[0],
                           wih + (size_t)n0 * II,
                           II, (unsigned)(GG - n0), II, 64);
        __builtin_amdgcn_s_wait_tensorcnt(0);
    }
#else
    for (int idx = tid; idx < 64 * II; idx += 256) {
        int ln = idx >> 8, k = idx & (II - 1);
        ldsW[idx] = wih[(size_t)(n0 + ln) * II + k];
    }
#endif
    __syncthreads();

    const size_t m0  = (size_t)blockIdx.x * 128 + wave * 16;
    const size_t row = m0 + (lane & 15);          // A-matrix M row (ISA 7.12.2)
    const int    lh  = lane >> 4;                 // lane half select

    v8f acc[4] = {v8f{0.f}, v8f{0.f}, v8f{0.f}, v8f{0.f}};

    for (int kt = 0; kt < II / 32; ++kt) {
        Frag a;
#pragma unroll
        for (int v = 0; v < 8; ++v) {
            // A 16x32 bf16 layout: K = 16*(v/4) + 8*(lane/16) + 2*(v%4)
            int k = kt * 32 + ((v >> 2) << 4) + (lh << 3) + ((v & 3) << 1);
            a.u[v] = *(const unsigned*)(xb + row * II + k);
        }
#pragma unroll
        for (int nt = 0; nt < 4; ++nt) {
            Frag b;
#pragma unroll
            for (int v = 0; v < 8; ++v) {
                // B 32x16 bf16 layout: K = 16*(v/4) + 8*(lane/16) + 2*(v%4), lane = N col
                int k  = kt * 32 + ((v >> 2) << 4) + (lh << 3) + ((v & 3) << 1);
                int ln = nt * 16 + (lane & 15);
                b.u[v] = *(const unsigned*)(&ldsW[ln * II + k]);
            }
            acc[nt] = __builtin_amdgcn_wmma_f32_16x16x32_bf16(
                false, a.v, false, b.v, (short)0, acc[nt], false, false);
        }
    }

#pragma unroll
    for (int nt = 0; nt < 4; ++nt) {
        int col = n0 + nt * 16 + (lane & 15);
        float bias = b_ih[col] + b_hh[col];
#pragma unroll
        for (int r = 0; r < 8; ++r) {
            size_t m = m0 + r + (lh << 3);        // C/D: M = r + 8*(lane/16)
            gates[m * GG + col] = acc[nt][r] + bias;
        }
    }
}

// ---------------- Phase 2: persistent recurrence, 16 WGs -------------------
__global__ __launch_bounds__(256) void lstm_recurrence(
    const float*  __restrict__ gates,  // [T*B, 4H]  (pre-activation x part)
    const __bf16* __restrict__ whh,    // [4H, H] bf16
    __bf16*       __restrict__ hbuf,   // [2][B*H] double-buffered h (bf16)
    float*        __restrict__ cbuf,   // [B*H] cell state
    float*        __restrict__ out,    // ys [T*B*H] ++ h_T [B*H] ++ c_T [B*H]
    unsigned*     __restrict__ bar)    // [T] arrive counters (zeroed by init)
{
    __shared__ __bf16 ldsW[64 * HH];   // 32 KB: W_hh rows for this WG's slice
    const int tid  = threadIdx.x;
    const int wave = tid >> 5;
    const int lane = tid & 31;
    const int lh   = lane >> 4;
    const int j0   = blockIdx.x * 16;  // hidden-column slice owned by this WG

    // Stage W_hh rows {g*256 + j0 .. +15} -> LDS rows g*16 .. +15 (all K).
    // Each gate group is a contiguous 16x256 2D tile -> 4 TDM descriptors.
#if __has_builtin(__builtin_amdgcn_tensor_load_to_lds)
    if (tid < 32) {   // wave 0 only
#pragma unroll
        for (int g = 0; g < 4; ++g) {
            tdm_load_2d_to_lds((unsigned)(uintptr_t)&ldsW[g * 16 * HH],
                               whh + (size_t)(g * 256 + j0) * HH,
                               HH, (unsigned)(GG - (g * 256 + j0)), HH, 16);
        }
        __builtin_amdgcn_s_wait_tensorcnt(0);
    }
#else
    for (int idx = tid; idx < 64 * HH; idx += 256) {
        int ln = idx >> 8, k = idx & (HH - 1);
        int ng = ((ln >> 4) << 8) + j0 + (ln & 15);
        ldsW[idx] = whh[(size_t)ng * HH + k];
    }
#endif
    __syncthreads();

    const int m0  = wave * 16;                 // batch-row tile for this wave
    const int row = m0 + (lane & 15);
    const int jj  = j0 + (lane & 15);

    for (int t = 0; t < TT; ++t) {
        const __bf16* hin  = hbuf + (size_t)(t & 1) * (BB * HH);
        __bf16*       hout = hbuf + (size_t)((t + 1) & 1) * (BB * HH);

        // Prefetch next timestep's gate segments (free: no counters). 32
        // lanes x 2 lines = all 64 cachelines this wave reads at t+1.
        if (t + 1 < TT) {
            const float* gnx =
                gates + ((size_t)(t + 1) * BB + m0 + (lane & 15)) * GG + j0;
            __builtin_prefetch(gnx + (size_t)lh * HH, 0, 3);
            __builtin_prefetch(gnx + (size_t)(lh + 2) * HH, 0, 3);
        }

        v8f acc[4] = {v8f{0.f}, v8f{0.f}, v8f{0.f}, v8f{0.f}};

        for (int kt = 0; kt < HH / 32; ++kt) {
            Frag a;
#pragma unroll
            for (int v = 0; v < 8; ++v) {
                int k = kt * 32 + ((v >> 2) << 4) + (lh << 3) + ((v & 3) << 1);
                a.u[v] = *(const unsigned*)(hin + row * HH + k);
            }
#pragma unroll
            for (int nt = 0; nt < 4; ++nt) {   // nt = gate group i,f,g,o
                Frag b;
#pragma unroll
                for (int v = 0; v < 8; ++v) {
                    int k  = kt * 32 + ((v >> 2) << 4) + (lh << 3) + ((v & 3) << 1);
                    int ln = nt * 16 + (lane & 15);
                    b.u[v] = *(const unsigned*)(&ldsW[ln * HH + k]);
                }
                acc[nt] = __builtin_amdgcn_wmma_f32_16x16x32_bf16(
                    false, a.v, false, b.v, (short)0, acc[nt], false, false);
            }
        }

        // LSTM cell update for (b = m0 + r + 8*lh, j = j0 + lane&15)
#pragma unroll
        for (int r = 0; r < 8; ++r) {
            int b = m0 + r + (lh << 3);
            const float* gx = gates + ((size_t)t * BB + b) * GG;
            float zi = acc[0][r] + gx[0 * HH + jj];
            float zf = acc[1][r] + gx[1 * HH + jj];
            float zg = acc[2][r] + gx[2 * HH + jj];
            float zo = acc[3][r] + gx[3 * HH + jj];
            float ig = sigmoid_f(zi);
            float fg = sigmoid_f(zf);
            float gg = tanh_f(zg);
            float og = sigmoid_f(zo);
            float cn = fg * cbuf[b * HH + jj] + ig * gg;
            float hn = og * tanh_f(cn);
            cbuf[b * HH + jj] = cn;
            out[((size_t)t * BB + b) * HH + jj] = hn;
            hout[b * HH + jj] = (__bf16)hn;
            if (t == TT - 1) {
                out[(size_t)TT * BB * HH + b * HH + jj]           = hn;
                out[(size_t)TT * BB * HH + BB * HH + b * HH + jj] = cn;
            }
        }

        // Device-wide barrier: all 16 WGs must publish hout before t+1.
        __syncthreads();
        if (tid == 0) {
            __threadfence();
            atomicAdd(&bar[t], 1u);
            while (__hip_atomic_load(&bar[t], __ATOMIC_ACQUIRE,
                                     __HIP_MEMORY_SCOPE_AGENT) < (unsigned)gridDim.x) {}
        }
        __syncthreads();
    }
}

// ---------------------------------------------------------------------------
extern "C" void kernel_launch(void* const* d_in, const int* in_sizes, int n_in,
                              void* d_out, int out_size, void* d_ws, size_t ws_size,
                              hipStream_t stream) {
    (void)in_sizes; (void)n_in; (void)out_size; (void)ws_size;
    const float* x    = (const float*)d_in[0];  // [T,B,I]
    const float* h0   = (const float*)d_in[1];  // [1,B,H]
    const float* c0   = (const float*)d_in[2];  // [1,B,H]
    const float* Wih  = (const float*)d_in[3];  // [4H,I]
    const float* Whh  = (const float*)d_in[4];  // [4H,H]
    const float* b_ih = (const float*)d_in[5];  // [4H]
    const float* b_hh = (const float*)d_in[6];  // [4H]
    float* out = (float*)d_out;

    // Workspace carve-up (~303 MB total), 256B-aligned slabs.
    char* ws = (char*)d_ws;
    size_t off = 0;
    auto carve = [&](size_t bytes) {
        void* p = ws + off;
        off = (off + bytes + 255) & ~(size_t)255;
        return p;
    };
    float*    gates = (float*)   carve((size_t)TT * BB * GG * 4); // 268 MB
    __bf16*   xb    = (__bf16*)  carve((size_t)TT * BB * II * 2); //  33 MB
    __bf16*   wihb  = (__bf16*)  carve((size_t)GG * II * 2);
    __bf16*   whhb  = (__bf16*)  carve((size_t)GG * HH * 2);
    __bf16*   hbuf  = (__bf16*)  carve((size_t)2 * BB * HH * 2);
    float*    cbuf  = (float*)   carve((size_t)BB * HH * 4);
    unsigned* bar   = (unsigned*)carve((size_t)TT * 4);

    lstm_cvt_bf16<<<2048, 256, 0, stream>>>(x,   xb,   TT * BB * II);
    lstm_cvt_bf16<<<64,   256, 0, stream>>>(Wih, wihb, GG * II);
    lstm_cvt_bf16<<<64,   256, 0, stream>>>(Whh, whhb, GG * HH);
    lstm_init<<<(BB * HH + 255) / 256, 256, 0, stream>>>(h0, c0, hbuf, cbuf, bar);

    dim3 g1(TT * BB / 128, GG / 64);  // (512, 16)
    lstm_gates_gemm<<<g1, 256, 0, stream>>>(xb, wihb, b_ih, b_hh, gates);

    lstm_recurrence<<<HH / 16, 256, 0, stream>>>(gates, whhb, hbuf, cbuf, out, bar);
}
